// MolecularGraphNeuralNetwork_14130442403945
// MI455X (gfx1250) — matrix operand order repaired and checked by reading, not compiled, not run on hardware
//
#include <hip/hip_runtime.h>
#include <hip/hip_bf16.h>
#include <math.h>

// ---------------------------------------------------------------------------
// MolecularGraphNeuralNetwork on MI455X (gfx1250, wave32, WMMA)
//   v = embed[fp]
//   3x: h = gelu(v @ Wf^T + bf); hs = h + A_blockdiag @ h; v = l2norm_rows(hs)
//   mol = meanpool(v); 2x: mol = gelu(mol @ Wo^T + bo); out = mol * mask
// All GEMMs via v_wmma_f32_16x16x32_bf16 (f32 accumulate).
// ---------------------------------------------------------------------------

typedef __attribute__((ext_vector_type(16))) __bf16 v16bf;
typedef __attribute__((ext_vector_type(8)))  __bf16 v8bf;
typedef __attribute__((ext_vector_type(8)))  float  v8f;

#define DIM     256
#define NATOMS  8192
#define MOLS    256
#define MATOMS  32

__device__ __forceinline__ __bf16 f2bf(float f) {
  unsigned u = __float_as_uint(f);
  unsigned r = u + 0x7FFFu + ((u >> 16) & 1u);   // round-to-nearest-even
  unsigned short s = (unsigned short)(r >> 16);
  __bf16 b;
  __builtin_memcpy(&b, &s, 2);
  return b;
}

__device__ __forceinline__ float gelu_tanh(float x) {
  float x3 = x * x * x;
  return 0.5f * x * (1.0f + tanhf(0.79788456080286536f * (x + 0.044715f * x3)));
}

// A-operand (16x32 bf16, MxK): lane m in 0..15 -> K = kb..kb+7 (elems 0..7)
// and K = kb+16..kb+23 (elems 8..15); lanes 16..31 use kb+8. Caller passes kb.
__device__ __forceinline__ v16bf load_a16(const __bf16* row, int kb) {
  v16bf r;
  *(v8bf*)&r         = *(const v8bf*)(row + kb);
  *(((v8bf*)&r) + 1) = *(const v8bf*)(row + kb + 16);
  return r;
}
// B-operand (32x16 bf16, KxN): lane n holds K = kb..kb+7 (elems 0..7) and
// K = kb+16..kb+23 (elems 8..15) of column n; lanes 16..31 use kb+8. `col` points at the
// column-contiguous (i.e. row of W / row of Ht) data.
__device__ __forceinline__ v16bf load_b16(const __bf16* col, int kb) {
  v16bf r;
  *(v8bf*)&r         = *(const v8bf*)(col + kb);
  *(((v8bf*)&r) + 1) = *(const v8bf*)(col + kb + 16);
  return r;
}

// ---------------------------------------------------------------------------
// Convert all f32 weights (3x Wf, 2x Wo) to bf16 once.
__global__ void convert_weights(const float* __restrict__ Wf,
                                const float* __restrict__ Wo,
                                __bf16* __restrict__ Wb) {
  int i = blockIdx.x * blockDim.x + threadIdx.x;     // 5*256*256 total
  if (i < 3 * DIM * DIM) Wb[i] = f2bf(Wf[i]);
  else                   Wb[i] = f2bf(Wo[i - 3 * DIM * DIM]);
}

// Embedding gather -> bf16 activations.
__global__ void gather_embed(const int* __restrict__ fp,
                             const float* __restrict__ emb,
                             __bf16* __restrict__ vB) {
  int row = blockIdx.x;            // atom
  int d   = threadIdx.x;           // feature
  int f   = fp[row];
  vB[(size_t)row * DIM + d] = f2bf(emb[(size_t)f * DIM + d]);
}

// ---------------------------------------------------------------------------
// out = gelu(A @ W^T + bias) [* mask], one 16x16 tile per wave32.
// A: [M,256] bf16 row-major, W: [256,256] bf16 row-major (so B = W^T and the
// per-lane B reads are W's rows, contiguous along K).
__global__ void gemm_bias_gelu(const __bf16* __restrict__ A,
                               const __bf16* __restrict__ W,
                               const float*  __restrict__ bias,
                               float*        __restrict__ outF,
                               __bf16*       __restrict__ outB,
                               const float*  __restrict__ mask,
                               int M) {
  int wave = (blockIdx.x * blockDim.x + threadIdx.x) >> 5;  // == tile index
  int lane = threadIdx.x & 31;
  int mt = wave >> 4;              // DIM/16 == 16 N-tiles
  int nt = wave & 15;
  int r  = lane & 15;
  int hi = lane >> 4;

  const __bf16* Arow = A + (size_t)(mt * 16 + r) * DIM;
  const __bf16* Wrow = W + (size_t)(nt * 16 + r) * DIM;

  v8f acc = {};
#pragma unroll
  for (int k0 = 0; k0 < DIM; k0 += 32) {
    v16bf a = load_a16(Arow, k0 + hi * 8);
    v16bf b = load_b16(Wrow, k0 + hi * 8);
    acc = __builtin_amdgcn_wmma_f32_16x16x32_bf16(false, a, false, b,
                                                  (short)0, acc, false, false);
  }

  int col  = nt * 16 + r;
  int row0 = mt * 16 + hi * 8;
  float bcol = bias[col];
#pragma unroll
  for (int i = 0; i < 8; ++i) {
    int row = row0 + i;
    float g = gelu_tanh(acc[i] + bcol);
    if (mask) g *= mask[row];
    outF[(size_t)row * DIM + col] = g;
    if (outB) outB[(size_t)row * DIM + col] = f2bf(g);
  }
}

// ---------------------------------------------------------------------------
// Per molecule: hs = h + A_block(32x32) @ h_block(32x256); v = hs / ||hs||_2.
// Only the diagonal 32x32 adjacency block is read (block-diagonal structure).
__global__ void adj_residual_norm(const float* __restrict__ adjacency,
                                  const float* __restrict__ hF,
                                  const __bf16* __restrict__ hB,
                                  float*  __restrict__ vF,
                                  __bf16* __restrict__ vB) {
  __shared__ __align__(32) __bf16 Ab[MATOMS * MATOMS];  // [m][k], 2 KB
  __shared__ __align__(32) __bf16 Ht[DIM * MATOMS];     // [n][k], 16 KB
  __shared__ float rowsq[MATOMS];

  int mol  = blockIdx.x;
  int tid  = threadIdx.x;          // 256 threads = 8 waves
  int base = mol * MATOMS;

  // Stage diagonal adjacency block (bf16-exact: values are 0/1).
  for (int i = tid; i < MATOMS * MATOMS; i += 256) {
    int m = i >> 5, k = i & 31;
    Ab[i] = f2bf(adjacency[(size_t)(base + m) * NATOMS + (base + k)]);
  }
  // Stage transposed H block: Ht[n][k] = h[base+k][n] (column-contiguous K).
  for (int i = tid; i < DIM * MATOMS; i += 256) {
    int n = i >> 5, k = i & 31;
    Ht[i] = hB[(size_t)(base + k) * DIM + n];
  }
  if (tid < MATOMS) rowsq[tid] = 0.0f;
  __syncthreads();

  int wave = tid >> 5, lane = tid & 31;
  int r = lane & 15, hi = lane >> 4;

  float hsv[4][8];                 // this wave's hs values (4 tiles x 8 rows)
#pragma unroll
  for (int i = 0; i < 4; ++i) {
    int t  = wave * 4 + i;         // 32 tiles: 2 M-tiles x 16 N-tiles
    int mt = t >> 4, nt = t & 15;
    v16bf a = load_a16(&Ab[(mt * 16 + r) * MATOMS], hi * 8);
    v16bf b = load_b16(&Ht[(nt * 16 + r) * MATOMS], hi * 8);
    v8f acc = {};
    acc = __builtin_amdgcn_wmma_f32_16x16x32_bf16(false, a, false, b,
                                                  (short)0, acc, false, false);
    int col  = nt * 16 + r;
    int row0 = mt * 16 + hi * 8;
#pragma unroll
    for (int j = 0; j < 8; ++j) {
      int row  = row0 + j;
      float hs = hF[(size_t)(base + row) * DIM + col] + acc[j];
      hsv[i][j] = hs;
      atomicAdd(&rowsq[row], hs * hs);    // ds_add_f32
    }
  }
  __syncthreads();

#pragma unroll
  for (int i = 0; i < 4; ++i) {
    int t  = wave * 4 + i;
    int mt = t >> 4, nt = t & 15;
    int col  = nt * 16 + r;
    int row0 = mt * 16 + hi * 8;
#pragma unroll
    for (int j = 0; j < 8; ++j) {
      int row = row0 + j;
      float nrm = fmaxf(sqrtf(rowsq[row]), 1e-12f);
      float val = hsv[i][j] / nrm;
      vF[(size_t)(base + row) * DIM + col] = val;
      vB[(size_t)(base + row) * DIM + col] = f2bf(val);
    }
  }
}

// Mean pool over the 32 atoms of each molecule.
__global__ void pool_mean(const float* __restrict__ vF,
                          __bf16* __restrict__ molB) {
  int b = blockIdx.x, d = threadIdx.x;
  float s = 0.0f;
#pragma unroll
  for (int a = 0; a < MATOMS; ++a)
    s += vF[((size_t)b * MATOMS + a) * DIM + d];
  molB[(size_t)b * DIM + d] = f2bf(s * (1.0f / MATOMS));
}

// ---------------------------------------------------------------------------
extern "C" void kernel_launch(void* const* d_in, const int* in_sizes, int n_in,
                              void* d_out, int out_size, void* d_ws, size_t ws_size,
                              hipStream_t stream) {
  (void)in_sizes; (void)n_in; (void)out_size; (void)ws_size;
  const int*   fp   = (const int*)  d_in[0];
  const float* adj  = (const float*)d_in[1];
  const float* mask = (const float*)d_in[2];
  const float* emb  = (const float*)d_in[3];
  const float* Wf   = (const float*)d_in[4];
  const float* bf   = (const float*)d_in[5];
  const float* Wo   = (const float*)d_in[6];
  const float* bo   = (const float*)d_in[7];

  char* p = (char*)d_ws;
  __bf16* Wb   = (__bf16*)p; p += (size_t)5 * DIM * DIM * sizeof(__bf16);
  __bf16* vB   = (__bf16*)p; p += (size_t)NATOMS * DIM * sizeof(__bf16);
  float*  vF   = (float*) p; p += (size_t)NATOMS * DIM * sizeof(float);
  float*  hF   = (float*) p; p += (size_t)NATOMS * DIM * sizeof(float);
  __bf16* hB   = (__bf16*)p; p += (size_t)NATOMS * DIM * sizeof(__bf16);
  __bf16* molB = (__bf16*)p; p += (size_t)MOLS * DIM * sizeof(__bf16);
  float*  m2F  = (float*) p; p += (size_t)MOLS * DIM * sizeof(float);
  __bf16* m2B  = (__bf16*)p; p += (size_t)MOLS * DIM * sizeof(__bf16);

  convert_weights<<<(5 * DIM * DIM) / 256, 256, 0, stream>>>(Wf, Wo, Wb);
  gather_embed<<<NATOMS, 256, 0, stream>>>(fp, emb, vB);

  for (int l = 0; l < 3; ++l) {
    gemm_bias_gelu<<<NATOMS / 8, 256, 0, stream>>>(
        vB, Wb + (size_t)l * DIM * DIM, bf + l * DIM, hF, hB, nullptr, NATOMS);
    adj_residual_norm<<<MOLS, 256, 0, stream>>>(adj, hF, hB, vF, vB);
  }

  pool_mean<<<MOLS, 256, 0, stream>>>(vF, molB);

  gemm_bias_gelu<<<MOLS / 8, 256, 0, stream>>>(
      molB, Wb + (size_t)3 * DIM * DIM, bo, m2F, m2B, nullptr, MOLS);
  gemm_bias_gelu<<<MOLS / 8, 256, 0, stream>>>(
      m2B, Wb + (size_t)4 * DIM * DIM, bo + DIM, (float*)d_out, nullptr, mask, MOLS);
}
